// MHA_65798898975512
// MI455X (gfx1250) — matrix operand order repaired and checked by reading, not compiled, not run on hardware
//
#include <hip/hip_runtime.h>
#include <hip/hip_bf16.h>

// ---------------------------------------------------------------------------
// MHA forward for gfx1250 (MI455X): f16 WMMA everywhere, f32 accumulation.
//   x:[4,2048,768] f32, w_q/k/v/o:[768,768] f32 -> out:[4,2048,768] f32
// Stages: convert->f16, fused QKV GEMM (Q scaled by 0.125, V stored
// transposed [b,h,d,t]), register-only flash attention (S^T trick so the
// softmax->A-fragment conversion is lane-local), output GEMM.
// ---------------------------------------------------------------------------

typedef _Float16 h8  __attribute__((ext_vector_type(8)));
typedef _Float16 v16h __attribute__((ext_vector_type(16)));
typedef float    v8f  __attribute__((ext_vector_type(8)));

#define HIDDEN 768
#define NHEADS 12
#define DHEAD  64
#define BB     4
#define TT     2048
#define MTOT   (BB * TT) // 8192

// A-fragment (16 rows x K=32, f16): lane L = row L%16, h = L/16.
// element e -> K = 8h+e (e<8), 16+8h+(e-8). Two 16B contiguous loads.
__device__ __forceinline__ v16h load_a_frag(const _Float16* p, int h) {
  h8 lo = *(const h8*)(p + 8 * h);
  h8 hi = *(const h8*)(p + 16 + 8 * h);
  v16h r;
#pragma unroll
  for (int i = 0; i < 8; ++i) { r[i] = lo[i]; r[8 + i] = hi[i]; }
  return r;
}

// B-fragment (K=32 x 16 cols, f16): lane L = col L%16, h = L/16.
// element e -> K = 8h+e (e<8), 16+8h+(e-8). Two 16B contiguous loads.
__device__ __forceinline__ v16h load_b_frag(const _Float16* p, int h) {
  h8 lo = *(const h8*)(p + 8 * h);
  h8 hi = *(const h8*)(p + 16 + 8 * h);
  v16h r;
#pragma unroll
  for (int i = 0; i < 8; ++i) { r[i] = lo[i]; r[8 + i] = hi[i]; }
  return r;
}

__device__ __forceinline__ v8f wmma_f16(v16h a, v16h b, v8f c) {
  return __builtin_amdgcn_wmma_f32_16x16x32_f16(
      /*neg_a=*/false, a, /*neg_b=*/false, b,
      /*c_mod=*/(short)0, c, /*reuse_a=*/false, /*reuse_b=*/false);
}

// ---------------------------------------------------------------------------
__global__ __launch_bounds__(256) void cvt_f32_to_f16(const float* __restrict__ in,
                                                      _Float16* __restrict__ out,
                                                      int n) {
  int i = blockIdx.x * blockDim.x + threadIdx.x;
  if (i < n) out[i] = (_Float16)in[i];
}

// ---------------------------------------------------------------------------
// Fused QKV projection: Y = x @ W^T. Each wave: 32 rows x 64 cols of one of
// {Q,K,V}. Q scaled by 1/sqrt(64); Q,K -> [b,h,t,d] f16; V -> [b,h,d,t] f16.
__global__ __launch_bounds__(256) void gemm_qkv(const _Float16* __restrict__ xh,
                                                const _Float16* __restrict__ wq,
                                                const _Float16* __restrict__ wk,
                                                const _Float16* __restrict__ wv,
                                                _Float16* __restrict__ qb,
                                                _Float16* __restrict__ kb,
                                                _Float16* __restrict__ vtb) {
  const int wid  = (blockIdx.x * blockDim.x + threadIdx.x) >> 5;
  const int lane = threadIdx.x & 31;
  const int h    = lane >> 4;
  const int l16  = lane & 15;

  const int NT = HIDDEN / 64;   // 12 col tiles
  const int MT = MTOT / 32;     // 256 row tiles
  const int mat = wid / (MT * NT);
  const int rem = wid % (MT * NT);
  const int m0  = (rem / NT) * 32;
  const int n0  = (rem % NT) * 64;
  const _Float16* w = (mat == 0) ? wq : ((mat == 1) ? wk : wv);

  v8f acc[2][4] = {};
  const _Float16* arow0 = xh + (m0 + l16) * HIDDEN;
  const _Float16* arow1 = xh + (m0 + 16 + l16) * HIDDEN;

  for (int k = 0; k < HIDDEN; k += 32) {
    v16h a0 = load_a_frag(arow0 + k, h);
    v16h a1 = load_a_frag(arow1 + k, h);
#pragma unroll
    for (int ni = 0; ni < 4; ++ni) {
      v16h b = load_b_frag(w + (n0 + ni * 16 + l16) * HIDDEN + k, h);
      acc[0][ni] = wmma_f16(a0, b, acc[0][ni]);
      acc[1][ni] = wmma_f16(a1, b, acc[1][ni]);
    }
  }

  const float scale = (mat == 0) ? 0.125f : 1.0f; // 1/sqrt(64) folded into Q
#pragma unroll
  for (int mi = 0; mi < 2; ++mi) {
#pragma unroll
    for (int ni = 0; ni < 4; ++ni) {
#pragma unroll
      for (int r = 0; r < 8; ++r) {
        int m = m0 + mi * 16 + r + 8 * h;   // D row
        int o = n0 + ni * 16 + l16;         // D col
        int b_ = m >> 11, t = m & (TT - 1);
        int head = o >> 6, d = o & (DHEAD - 1);
        _Float16 v = (_Float16)(acc[mi][ni][r] * scale);
        if (mat == 0)      qb[(((b_ * NHEADS + head) * TT) + t) * DHEAD + d] = v;
        else if (mat == 1) kb[(((b_ * NHEADS + head) * TT) + t) * DHEAD + d] = v;
        else               vtb[((b_ * NHEADS + head) * DHEAD + d) * TT + t] = v;
      }
    }
  }
}

// ---------------------------------------------------------------------------
// Flash attention, one wave = 32 queries of one (b,head). Computes S^T
// (A = K rows, B = Q^T) so each lane owns one query column; the exp'd
// probabilities map lane-locally onto the A-fragment layout for ctx = P @ V.
__global__ __launch_bounds__(256) void attn_flash(const _Float16* __restrict__ qb,
                                                  const _Float16* __restrict__ kb,
                                                  const _Float16* __restrict__ vtb,
                                                  _Float16* __restrict__ ctxh) {
  const int wid  = (blockIdx.x * blockDim.x + threadIdx.x) >> 5;
  const int lane = threadIdx.x & 31;
  const int h    = lane >> 4;
  const int l16  = lane & 15;

  const int bh = wid >> 6;        // 64 query tiles of 32 per (b,head)
  const int q0 = (wid & 63) * 32;
  const _Float16* Q  = qb  + (size_t)bh * TT * DHEAD;
  const _Float16* K  = kb  + (size_t)bh * TT * DHEAD;
  const _Float16* Vt = vtb + (size_t)bh * DHEAD * TT;

  // Q as B-matrix: column = query (lane), d = 8h+e (e<8), 16+8h+(e-8).
  v16h qf[2][2];
#pragma unroll
  for (int qt = 0; qt < 2; ++qt)
#pragma unroll
    for (int c = 0; c < 2; ++c)
      qf[qt][c] = load_b_frag(Q + (q0 + qt * 16 + l16) * DHEAD + c * 32, h);

  v8f acc[2][4] = {};
  float m_run[2] = {-1e30f, -1e30f};
  float l_run[2] = {0.0f, 0.0f};
  const float L2E = 1.44269504088896f;

  for (int kc = 0; kc < TT; kc += 32) {
    // K tiles as A-matrix (rows = keys), d split in two K=32 chunks.
    v16h ka00 = load_a_frag(K + (kc + l16) * DHEAD, h);
    v16h ka01 = load_a_frag(K + (kc + l16) * DHEAD + 32, h);
    v16h ka10 = load_a_frag(K + (kc + 16 + l16) * DHEAD, h);
    v16h ka11 = load_a_frag(K + (kc + 16 + l16) * DHEAD + 32, h);
    // V as B-matrix (rows = keys, cols = d): contiguous along t in Vt.
    v16h vb[4];
#pragma unroll
    for (int j = 0; j < 4; ++j)
      vb[j] = load_b_frag(Vt + (16 * j + l16) * TT + kc, h);

#pragma unroll
    for (int qt = 0; qt < 2; ++qt) {
      v8f s0 = {}, s1 = {};
      s0 = wmma_f16(ka00, qf[qt][0], s0);
      s0 = wmma_f16(ka01, qf[qt][1], s0);   // keys kc..kc+15   x queries
      s1 = wmma_f16(ka10, qf[qt][0], s1);
      s1 = wmma_f16(ka11, qf[qt][1], s1);   // keys kc+16..kc+31 x queries

      // Online softmax: lane owns query l16; its 16 keys are {8h+r, 16+8h+r}.
      float cmax = -1e30f;
#pragma unroll
      for (int e = 0; e < 8; ++e)
        cmax = fmaxf(cmax, fmaxf(s0[e], s1[e]));
      cmax = fmaxf(cmax, __shfl_xor(cmax, 16, 32)); // partner half, same query
      float mnew  = fmaxf(m_run[qt], cmax);
      float rescale = exp2f((m_run[qt] - mnew) * L2E);
      m_run[qt] = mnew;

      v16h p;
      float psum = 0.0f;
#pragma unroll
      for (int e = 0; e < 8; ++e) {
        float p0 = exp2f((s0[e] - mnew) * L2E);  // A-frag element e
        float p1 = exp2f((s1[e] - mnew) * L2E);  // A-frag element 8+e
        p[e] = (_Float16)p0;
        p[8 + e] = (_Float16)p1;
        psum += p0 + p1;
      }
      l_run[qt] = l_run[qt] * rescale + psum;

      // Accumulator rows are queries r+8h: broadcast that query's rescale.
      float sc[8];
#pragma unroll
      for (int r = 0; r < 8; ++r) sc[r] = __shfl(rescale, 8 * h + r, 32);
#pragma unroll
      for (int j = 0; j < 4; ++j) {
#pragma unroll
        for (int r = 0; r < 8; ++r) acc[qt][j][r] *= sc[r];
        acc[qt][j] = wmma_f16(p, vb[j], acc[qt][j]); // ctx += P @ V
      }
    }
  }

  // Epilogue: normalize by row sums and store ctx f16 as [b,t,hidden].
  const int b_ = bh / NHEADS, head = bh % NHEADS;
#pragma unroll
  for (int qt = 0; qt < 2; ++qt) {
    float lt  = l_run[qt] + __shfl_xor(l_run[qt], 16, 32);
    float inv = 1.0f / lt;
    float invr[8];
#pragma unroll
    for (int r = 0; r < 8; ++r) invr[r] = __shfl(inv, 8 * h + r, 32);
#pragma unroll
    for (int j = 0; j < 4; ++j) {
#pragma unroll
      for (int r = 0; r < 8; ++r) {
        int q = q0 + qt * 16 + r + 8 * h;
        int d = 16 * j + l16;
        ctxh[((size_t)(b_ * TT + q)) * HIDDEN + head * DHEAD + d] =
            (_Float16)(acc[qt][j][r] * invr[r]);
      }
    }
  }
}

// ---------------------------------------------------------------------------
// Output projection: out = ctx @ w_o^T, f32 output. 32x64 tile per wave.
__global__ __launch_bounds__(256) void gemm_out(const _Float16* __restrict__ ctxh,
                                                const _Float16* __restrict__ wo,
                                                float* __restrict__ out) {
  const int wid  = (blockIdx.x * blockDim.x + threadIdx.x) >> 5;
  const int lane = threadIdx.x & 31;
  const int h    = lane >> 4;
  const int l16  = lane & 15;

  const int NT = HIDDEN / 64; // 12
  const int m0 = (wid / NT) * 32;
  const int n0 = (wid % NT) * 64;

  v8f acc[2][4] = {};
  const _Float16* arow0 = ctxh + (m0 + l16) * HIDDEN;
  const _Float16* arow1 = ctxh + (m0 + 16 + l16) * HIDDEN;

  for (int k = 0; k < HIDDEN; k += 32) {
    v16h a0 = load_a_frag(arow0 + k, h);
    v16h a1 = load_a_frag(arow1 + k, h);
#pragma unroll
    for (int ni = 0; ni < 4; ++ni) {
      v16h b = load_b_frag(wo + (n0 + ni * 16 + l16) * HIDDEN + k, h);
      acc[0][ni] = wmma_f16(a0, b, acc[0][ni]);
      acc[1][ni] = wmma_f16(a1, b, acc[1][ni]);
    }
  }

#pragma unroll
  for (int mi = 0; mi < 2; ++mi)
#pragma unroll
    for (int ni = 0; ni < 4; ++ni)
#pragma unroll
      for (int r = 0; r < 8; ++r) {
        int m = m0 + mi * 16 + r + 8 * h;
        int o = n0 + ni * 16 + l16;
        out[(size_t)m * HIDDEN + o] = acc[mi][ni][r];
      }
}

// ---------------------------------------------------------------------------
extern "C" void kernel_launch(void* const* d_in, const int* in_sizes, int n_in,
                              void* d_out, int out_size, void* d_ws, size_t ws_size,
                              hipStream_t stream) {
  const float* x  = (const float*)d_in[0];
  const float* wq = (const float*)d_in[1];
  const float* wk = (const float*)d_in[2];
  const float* wv = (const float*)d_in[3];
  const float* wo = (const float*)d_in[4];
  float* out = (float*)d_out;

  const size_t NX = (size_t)MTOT * HIDDEN;      // 6,291,456
  const size_t NW = (size_t)HIDDEN * HIDDEN;    //   589,824

  _Float16* ws = (_Float16*)d_ws;
  _Float16* xh  = ws;                 // [8192,768]
  _Float16* wqh = xh  + NX;           // [768,768]
  _Float16* wkh = wqh + NW;
  _Float16* wvh = wkh + NW;
  _Float16* woh = wvh + NW;
  _Float16* qbf = woh + NW;           // [b,h,t,d]
  _Float16* kbf = qbf + NX;           // [b,h,t,d]
  _Float16* vtf = kbf + NX;           // [b,h,d,t]
  _Float16* ctx = vtf + NX;           // [b,t,hidden]
  // total: 5*NX + 4*NW halves = ~67.6 MB of d_ws

  // Stage 1: convert to f16.
  cvt_f32_to_f16<<<(int)((NX + 255) / 256), 256, 0, stream>>>(x,  xh,  (int)NX);
  cvt_f32_to_f16<<<(int)((NW + 255) / 256), 256, 0, stream>>>(wq, wqh, (int)NW);
  cvt_f32_to_f16<<<(int)((NW + 255) / 256), 256, 0, stream>>>(wk, wkh, (int)NW);
  cvt_f32_to_f16<<<(int)((NW + 255) / 256), 256, 0, stream>>>(wv, wvh, (int)NW);
  cvt_f32_to_f16<<<(int)((NW + 255) / 256), 256, 0, stream>>>(wo, woh, (int)NW);

  // Stage 2: fused QKV projections. waves = 3 * 256 * 12 = 9216 -> 1152 blocks.
  gemm_qkv<<<1152, 256, 0, stream>>>(xh, wqh, wkh, wvh, qbf, kbf, vtf);

  // Stage 3: flash attention. waves = 4*12*(2048/32) = 3072 -> 384 blocks.
  attn_flash<<<384, 256, 0, stream>>>(qbf, kbf, vtf, ctx);

  // Stage 4: output projection. waves = 256*12 = 3072 -> 384 blocks.
  gemm_out<<<384, 256, 0, stream>>>(ctx, woh, out);
}
